// ConformerEncoderLayer_42047729828097
// MI455X (gfx1250) — compile-verified, matrix operand order checked by reading, not run on hardware
//
#include <hip/hip_runtime.h>

// ---------------------------------------------------------------------------
// Conformer encoder layer for MI455X (gfx1250, wave32, WMMA + TDM).
// GEMMs: v_wmma_f32_16x16x32_bf16, LDS tiles staged by the Tensor Data Mover
// (double-buffered, TENSORcnt-synchronized) when the TDM builtin is available.
// ---------------------------------------------------------------------------

typedef __bf16 bf16;
typedef __attribute__((ext_vector_type(16))) __bf16 v16bf;
typedef __attribute__((ext_vector_type(8)))  __bf16 v8bf;
typedef __attribute__((ext_vector_type(8)))  float  v8f;

#define CB 16
#define CL 512
#define CD 512
#define CH 8
#define CHD 64
#define CFF 2048
#define CBL (CB * CL)   // 8192 tokens

#if defined(__HIP_DEVICE_COMPILE__) && __has_builtin(__builtin_amdgcn_tensor_load_to_lds) && \
    __has_builtin(__builtin_amdgcn_s_wait_tensorcnt)
#define USE_TDM 1
#else
#define USE_TDM 0
#endif

// ---------------------------------------------------------------------------
// WMMA helpers
// ---------------------------------------------------------------------------
__device__ inline v8f wmma_bf16(v16bf a, v16bf b, v8f c) {
  // D = A(16x32) * B(32x16) + C(16x16 f32)
  return __builtin_amdgcn_wmma_f32_16x16x32_bf16(
      /*neg_a=*/false, a, /*neg_b=*/false, b,
      /*c_mod=*/(short)0, c, /*reuse_a=*/false, /*reuse_b=*/false);
}

// 16-bit A/B fragment: lane<16 holds K[0..7] and K[16..23] of row (lane&15),
// lane>=16 holds K[8..15] and K[24..31] of the same row. Two 16-byte loads.
__device__ inline v16bf ldfrag(const bf16* __restrict__ p, int rstride, int lane) {
  const int r = lane & 15;
  const int half = (lane >> 4) << 3;       // 0 or 8 elements
  const bf16* q = p + r * rstride + half;
  union { v16bf v; v8bf h[2]; } u;
  u.h[0] = *(const v8bf*)(q);
  u.h[1] = *(const v8bf*)(q + 16);
  return u.v;
}

__device__ inline float wave_red_sum(float v) {
#pragma unroll
  for (int o = 16; o; o >>= 1) v += __shfl_xor(v, o, 32);
  return v;
}

// ---------------------------------------------------------------------------
// TDM: 2-D tile load Global -> LDS, rows of 32 bf16 (64B), DMA-inserted LDS
// padding of 16B per row => LDS row stride 40 bf16 (matches ldfrag).
// Descriptor packing per CDNA5 ISA.
// ---------------------------------------------------------------------------
#if USE_TDM
typedef unsigned int u32x4 __attribute__((ext_vector_type(4)));
typedef int i32x4 __attribute__((ext_vector_type(4)));
typedef int i32x8 __attribute__((ext_vector_type(8)));

__device__ inline unsigned lds_off(const void* p) {
  return (unsigned)(size_t)(const __attribute__((address_space(3))) void*)p;
}

__device__ inline void tdm_load_tile(unsigned lds_byte_addr, const void* gsrc,
                                     unsigned rows, unsigned row_stride_elems) {
  const unsigned long long ga = (unsigned long long)(size_t)gsrc;
  u32x4 g0;
  g0[0] = 1u;                                   // count=1 (valid), user mode
  g0[1] = lds_byte_addr;                        // lds_addr [63:32]
  g0[2] = (unsigned)(ga & 0xffffffffu);         // global_addr lo
  g0[3] = (unsigned)(ga >> 32) | (2u << 30);    // global_addr hi | type=2
  const unsigned td0 = 32u;                     // tensor_dim0 (elements)
  const unsigned td1 = rows;                    // tensor_dim1
  i32x8 g1;
  // data_size=1 (2B) | pad_enable | pad_interval=16DW (code 3) | pad_amount=4DW (code 3)
  g1[0] = (int)((1u << 16) | (1u << 20) | (3u << 22) | (3u << 25));
  g1[1] = (int)((td0 & 0xffffu) << 16);                       // dim0[15:0] @ [63:48]
  g1[2] = (int)((td0 >> 16) | ((td1 & 0xffffu) << 16));       // dim0[31:16] | dim1[15:0]
  g1[3] = (int)((td1 >> 16) | (32u << 16));                   // dim1[31:16] | tile_dim0=32
  g1[4] = (int)rows;                                          // tile_dim1=rows, tile_dim2=0
  g1[5] = (int)row_stride_elems;                              // tensor_dim0_stride lo32
  g1[6] = 0;                                                  // stride hi | dim1_stride lo
  g1[7] = 0;
  i32x4 z4 = {};
  i32x8 z8 = {};
  // 6-arg flavor (clang-23 / therock-10.0): (g0, g1, g2, g3, g4, cpol)
  __builtin_amdgcn_tensor_load_to_lds(g0, g1, z4, z4, z8, 0);
}
#endif

// ---------------------------------------------------------------------------
// fp32 -> bf16 weight conversion
// ---------------------------------------------------------------------------
__global__ __launch_bounds__(256) void cvt_bf16_k(const float* __restrict__ s,
                                                  bf16* __restrict__ d, int n) {
  int i = blockIdx.x * 256 + threadIdx.x;
  if (i < n) d[i] = (bf16)s[i];
}

// ---------------------------------------------------------------------------
// LayerNorm over D=512, one block (256 thr) per token.
// OM=0: bf16 out; OM=1: f32 out.
// ---------------------------------------------------------------------------
template <int OM>
__global__ __launch_bounds__(256)
void ln_k(const float* __restrict__ X, const float* __restrict__ g,
          const float* __restrict__ bb, float* __restrict__ outf,
          bf16* __restrict__ outb) {
  __shared__ float red[8];
  __shared__ float sstat[2];
  const int row = blockIdx.x, tid = threadIdx.x;
  const float* xr = X + (size_t)row * CD;
  float a = xr[tid], c = xr[tid + 256];
  float s = wave_red_sum(a + c);
  if ((tid & 31) == 0) red[tid >> 5] = s;
  __syncthreads();
  if (tid == 0) { float t = 0.f; for (int i = 0; i < 8; i++) t += red[i]; sstat[0] = t * (1.f / CD); }
  __syncthreads();
  const float mu = sstat[0];
  const float da = a - mu, dc = c - mu;
  float v = wave_red_sum(da * da + dc * dc);
  if ((tid & 31) == 0) red[tid >> 5] = v;
  __syncthreads();
  if (tid == 0) { float t = 0.f; for (int i = 0; i < 8; i++) t += red[i]; sstat[1] = rsqrtf(t * (1.f / CD) + 1e-5f); }
  __syncthreads();
  const float rs = sstat[1];
  const float y0 = da * rs * g[tid] + bb[tid];
  const float y1 = dc * rs * g[tid + 256] + bb[tid + 256];
  if (OM == 0) {
    outb[(size_t)row * CD + tid] = (bf16)y0;
    outb[(size_t)row * CD + tid + 256] = (bf16)y1;
  } else {
    outf[(size_t)row * CD + tid] = y0;
    outf[(size_t)row * CD + tid + 256] = y1;
  }
}

// LN2 + RoPE: emits plain bf16 (for V) and rotated bf16 (for Q/K).
__global__ __launch_bounds__(256)
void ln_rope_k(const float* __restrict__ X, const float* __restrict__ g,
               const float* __restrict__ bb, bf16* __restrict__ outh,
               bf16* __restrict__ outr) {
  __shared__ float red[8];
  __shared__ float sstat[2];
  const int row = blockIdx.x, tid = threadIdx.x;
  const float* xr = X + (size_t)row * CD;
  float a = xr[tid], c = xr[tid + 256];
  float s = wave_red_sum(a + c);
  if ((tid & 31) == 0) red[tid >> 5] = s;
  __syncthreads();
  if (tid == 0) { float t = 0.f; for (int i = 0; i < 8; i++) t += red[i]; sstat[0] = t * (1.f / CD); }
  __syncthreads();
  const float mu = sstat[0];
  const float da = a - mu, dc = c - mu;
  float v = wave_red_sum(da * da + dc * dc);
  if ((tid & 31) == 0) red[tid >> 5] = v;
  __syncthreads();
  if (tid == 0) { float t = 0.f; for (int i = 0; i < 8; i++) t += red[i]; sstat[1] = rsqrtf(t * (1.f / CD) + 1e-5f); }
  __syncthreads();
  const float rs = sstat[1];
  const float y0 = da * rs * g[tid] + bb[tid];
  const float y1 = dc * rs * g[tid + 256] + bb[tid + 256];
  outh[(size_t)row * CD + tid] = (bf16)y0;
  outh[(size_t)row * CD + tid + 256] = (bf16)y1;
  // RoPE: pair (tid, tid+256); inv-freq rounded through bf16 like reference.
  const int l = row & (CL - 1);
  const float sc = (float)(bf16)__powf(10000.f, -(float)tid * (1.f / 256.f));
  const float th = (float)l * sc;
  const float sn = __sinf(th), cs = __cosf(th);
  outr[(size_t)row * CD + tid] = (bf16)(y0 * cs - y1 * sn);
  outr[(size_t)row * CD + tid + 256] = (bf16)(y1 * cs + y0 * sn);
}

// ---------------------------------------------------------------------------
// WMMA GEMM:  C[M,N] = epi( A[M,K](bf16) * W[N,K]^T(bf16) + bias )
// Workgroup tile 128x64, 8 waves as 4(M)x2(N), wave tile 32x32 (2x2 wmma).
// TDM path: wave0 DMAs the A tile, wave1 the W tile, double-buffered in LDS.
// ---------------------------------------------------------------------------
enum { EPI_RELU_BF16 = 0, EPI_RES = 1, EPI_BF16 = 2, EPI_VT = 3, EPI_F32 = 4 };

#define LDS_STR 40  // 32 + 8 bf16 pad (TDM pad_amount); 16B-aligned rows

template <int EPI>
__global__ __launch_bounds__(256, 2)
void gemm_bf16_k(const bf16* __restrict__ A, const bf16* __restrict__ W,
                 const float* __restrict__ bias, float* __restrict__ outf,
                 bf16* __restrict__ outb, float scale, int M, int N, int K,
                 int Lt) {
  __shared__ __align__(16) bf16 smem[2 * (128 + 64) * LDS_STR];
  const int tid = threadIdx.x, lane = tid & 31, wid = tid >> 5;
  const int m0 = blockIdx.x * 128, n0 = blockIdx.y * 64;
  const int wm = (wid & 3) * 32, wn = (wid >> 2) * 32;
  v8f acc00 = {}, acc01 = {}, acc10 = {}, acc11 = {};

#if USE_TDM
  const int NT = K / 32;
  const unsigned ldsA0 = lds_off(smem);
  const unsigned bufBytes = 192u * LDS_STR * 2u;
  // prologue: DMA tile 0 into buffer 0
  if (wid == 0)
    tdm_load_tile(ldsA0, &A[(size_t)m0 * K], 128, (unsigned)K);
  else if (wid == 1)
    tdm_load_tile(ldsA0 + 128u * LDS_STR * 2u, &W[(size_t)n0 * K], 64, (unsigned)K);
  for (int t = 0; t < NT; ++t) {
    const int cur = t & 1;
    if (t + 1 < NT) {
      const unsigned bo = (unsigned)(cur ^ 1) * bufBytes;
      if (wid == 0)
        tdm_load_tile(ldsA0 + bo, &A[(size_t)m0 * K + (t + 1) * 32], 128, (unsigned)K);
      else if (wid == 1)
        tdm_load_tile(ldsA0 + bo + 128u * LDS_STR * 2u,
                      &W[(size_t)n0 * K + (t + 1) * 32], 64, (unsigned)K);
      __builtin_amdgcn_s_wait_tensorcnt(1);   // retire tile t, keep t+1 in flight
    } else {
      __builtin_amdgcn_s_wait_tensorcnt(0);
    }
    __syncthreads();
    const bf16* cA = smem + cur * (192 * LDS_STR);
    const bf16* cB = cA + 128 * LDS_STR;
    v16bf a0 = ldfrag(&cA[wm * LDS_STR], LDS_STR, lane);
    v16bf a1 = ldfrag(&cA[(wm + 16) * LDS_STR], LDS_STR, lane);
    v16bf b0 = ldfrag(&cB[wn * LDS_STR], LDS_STR, lane);
    v16bf b1 = ldfrag(&cB[(wn + 16) * LDS_STR], LDS_STR, lane);
    acc00 = wmma_bf16(a0, b0, acc00);
    acc01 = wmma_bf16(a0, b1, acc01);
    acc10 = wmma_bf16(a1, b0, acc10);
    acc11 = wmma_bf16(a1, b1, acc11);
    __syncthreads();   // buffer cur is DMA-overwritten next iteration
  }
#else
  bf16* sA = smem;
  bf16* sB = smem + 128 * LDS_STR;
  for (int k0 = 0; k0 < K; k0 += 32) {
    __syncthreads();
    {
      int cidx = tid * 2;
      int r = cidx >> 2, cc = (cidx & 3) * 8;
      *(v8bf*)&sA[r * LDS_STR + cc] = *(const v8bf*)&A[(size_t)(m0 + r) * K + k0 + cc];
      cidx++;
      r = cidx >> 2; cc = (cidx & 3) * 8;
      *(v8bf*)&sA[r * LDS_STR + cc] = *(const v8bf*)&A[(size_t)(m0 + r) * K + k0 + cc];
    }
    {
      int r = tid >> 2, cc = (tid & 3) * 8;
      *(v8bf*)&sB[r * LDS_STR + cc] = *(const v8bf*)&W[(size_t)(n0 + r) * K + k0 + cc];
    }
    __syncthreads();
    v16bf a0 = ldfrag(&sA[wm * LDS_STR], LDS_STR, lane);
    v16bf a1 = ldfrag(&sA[(wm + 16) * LDS_STR], LDS_STR, lane);
    v16bf b0 = ldfrag(&sB[wn * LDS_STR], LDS_STR, lane);
    v16bf b1 = ldfrag(&sB[(wn + 16) * LDS_STR], LDS_STR, lane);
    acc00 = wmma_bf16(a0, b0, acc00);
    acc01 = wmma_bf16(a0, b1, acc01);
    acc10 = wmma_bf16(a1, b0, acc10);
    acc11 = wmma_bf16(a1, b1, acc11);
  }
#endif

  // Epilogue. Acc layout: lane 0-15 col n=lane, rows i; lane 16-31 rows i+8.
  const int moff = (lane >> 4) << 3;
  const int ncol = lane & 15;
#pragma unroll
  for (int t = 0; t < 4; t++) {
    const v8f acc = (t == 0) ? acc00 : (t == 1) ? acc01 : (t == 2) ? acc10 : acc11;
    const int mb = m0 + wm + ((t >> 1) << 4);
    const int n = n0 + wn + ((t & 1) << 4) + ncol;
    const float bv = bias ? bias[n] : 0.f;
#pragma unroll
    for (int i = 0; i < 8; i++) {
      const int m = mb + moff + i;
      const float v = acc[i] + bv;
      if (EPI == EPI_RELU_BF16) {
        outb[(size_t)m * N + n] = (bf16)fmaxf(v, 0.f);
      } else if (EPI == EPI_RES) {
        outf[(size_t)m * N + n] += scale * v;
      } else if (EPI == EPI_BF16) {
        outb[(size_t)m * N + n] = (bf16)v;
      } else if (EPI == EPI_F32) {
        outf[(size_t)m * N + n] = v;
      } else {  // EPI_VT: store V transposed as [B, N(=D), L]
        const int bb2 = m / Lt, l = m % Lt;
        outb[((size_t)bb2 * N + n) * Lt + l] = (bf16)v;
      }
    }
  }
}

// ---------------------------------------------------------------------------
// Attention: one block per (b, h, 16-row q tile).
// Phase1 S=Q*K^T (WMMA, 1/8 scale) -> LDS; Phase2 exact softmax; Phase3 P*V^T.
// ---------------------------------------------------------------------------
__global__ __launch_bounds__(256)
void attn_k(const bf16* __restrict__ Q, const bf16* __restrict__ Km,
            const bf16* __restrict__ Vt, bf16* __restrict__ O) {
  __shared__ float S[16 * CL];
  __shared__ bf16 P[16 * CL];
  __shared__ float red[256];
  const int tid = threadIdx.x, lane = tid & 31, wid = tid >> 5;
  const int bh = blockIdx.x >> 5;   // 32 q-tiles per (b,h)
  const int qt = blockIdx.x & 31;
  const int b = bh >> 3, h = bh & 7;
  const int q0 = qt * 16;
  const int moff = (lane >> 4) << 3;

  // ---- Phase 1: scores ----
  const bf16* qbase = Q + ((size_t)(b * CL + q0)) * CD + h * CHD;
#pragma unroll
  for (int ti = 0; ti < 4; ti++) {
    const int ct = wid * 4 + ti;  // key tile (16 keys)
    v8f acc = {};
    const bf16* kbase = Km + ((size_t)(b * CL + ct * 16)) * CD + h * CHD;
#pragma unroll
    for (int kk = 0; kk < CHD; kk += 32) {
      v16bf a = ldfrag(qbase + kk, CD, lane);
      v16bf bm = ldfrag(kbase + kk, CD, lane);
      acc = wmma_bf16(a, bm, acc);
    }
    const int n = ct * 16 + (lane & 15);
#pragma unroll
    for (int i = 0; i < 8; i++) S[(moff + i) * CL + n] = acc[i] * 0.125f;
  }
  __syncthreads();

  // ---- Phase 2: softmax (16 threads per row) ----
  const int row = tid >> 4, sub = tid & 15;
  float mx = -3.0e38f;
  for (int c = sub; c < CL; c += 16) mx = fmaxf(mx, S[row * CL + c]);
  red[tid] = mx;
  __syncthreads();
  if (sub == 0) {
    float m = red[row * 16];
    for (int i = 1; i < 16; i++) m = fmaxf(m, red[row * 16 + i]);
    red[row * 16] = m;
  }
  __syncthreads();
  mx = red[row * 16];
  float sm = 0.f;
  for (int c = sub; c < CL; c += 16) {
    const float e = __expf(S[row * CL + c] - mx);
    S[row * CL + c] = e;
    sm += e;
  }
  __syncthreads();
  red[tid] = sm;
  __syncthreads();
  if (sub == 0) {
    float t = 0.f;
    for (int i = 0; i < 16; i++) t += red[row * 16 + i];
    red[row * 16] = t;
  }
  __syncthreads();
  const float inv = 1.f / red[row * 16];
  for (int c = sub; c < CL; c += 16) P[row * CL + c] = (bf16)(S[row * CL + c] * inv);
  __syncthreads();

  // ---- Phase 3: O = P * V  (V stored transposed [B, D, L]) ----
  if (wid < 4) {  // wave-uniform branch; EXEC stays all-ones inside
    v8f acc = {};
    const bf16* vb = Vt + ((size_t)b * CD + h * CHD + wid * 16) * CL;
    for (int kk = 0; kk < CL; kk += 32) {
      v16bf a = ldfrag(&P[kk], CL, lane);
      v16bf bm = ldfrag(vb + kk, CL, lane);
      acc = wmma_bf16(a, bm, acc);
    }
    const int n = h * CHD + wid * 16 + (lane & 15);
#pragma unroll
    for (int i = 0; i < 8; i++) {
      const int m = q0 + moff + i;
      O[((size_t)(b * CL + m)) * CD + n] = (bf16)acc[i];
    }
  }
}

// ---------------------------------------------------------------------------
// Conv-module elementwise pieces
// ---------------------------------------------------------------------------
__global__ __launch_bounds__(256)
void glu_k(const float* __restrict__ G, float* __restrict__ out) {
  const size_t i = (size_t)blockIdx.x * 256 + threadIdx.x;  // over BL*D
  const size_t bl = i / CD, d = i % CD;
  const float a = G[bl * (2 * CD) + d];
  const float g = G[bl * (2 * CD) + CD + d];
  out[i] = a * (1.f / (1.f + __expf(-g)));
}

__global__ __launch_bounds__(256)
void dwconv_k(const float* __restrict__ in, const float* __restrict__ w,
              float* __restrict__ out) {
  const size_t i = (size_t)blockIdx.x * 256 + threadIdx.x;  // over BL*D
  const int d = (int)(i % CD);
  const size_t bl = i / CD;
  const int b = (int)(bl / CL), l = (int)(bl % CL);
  float acc = 0.f;
#pragma unroll
  for (int t = 0; t < 31; t++) {
    const int ll = l + t - 15;
    if (ll >= 0 && ll < CL)
      acc += in[((size_t)(b * CL + ll)) * CD + d] * w[d * 31 + t];
  }
  out[i] = acc;
}

__global__ __launch_bounds__(256)
void bnstats_k(const float* __restrict__ h, float* __restrict__ stats) {
  __shared__ float reds[8], redq[8];
  const int d = blockIdx.x, tid = threadIdx.x;
  float s = 0.f, q = 0.f;
  for (int r = tid; r < CBL; r += 256) {
    const float v = h[(size_t)r * CD + d];
    s += v;
    q += v * v;
  }
  s = wave_red_sum(s);
  q = wave_red_sum(q);
  if ((tid & 31) == 0) { reds[tid >> 5] = s; redq[tid >> 5] = q; }
  __syncthreads();
  if (tid == 0) {
    float ts = 0.f, tq = 0.f;
    for (int i = 0; i < 8; i++) { ts += reds[i]; tq += redq[i]; }
    const float mu = ts * (1.f / CBL);
    stats[d] = mu;
    stats[CD + d] = tq * (1.f / CBL) - mu * mu;
  }
}

__global__ __launch_bounds__(256)
void bnsilu_k(const float* __restrict__ h, const float* __restrict__ stats,
              const float* __restrict__ g, const float* __restrict__ b,
              bf16* __restrict__ out) {
  const size_t i = (size_t)blockIdx.x * 256 + threadIdx.x;  // over BL*D
  const int d = (int)(i % CD);
  const float y = (h[i] - stats[d]) * rsqrtf(stats[CD + d] + 1e-5f) * g[d] + b[d];
  out[i] = (bf16)(y * (1.f / (1.f + __expf(-y))));
}

// ---------------------------------------------------------------------------
// Host-side orchestration
// ---------------------------------------------------------------------------
extern "C" void kernel_launch(void* const* d_in, const int* in_sizes, int n_in,
                              void* d_out, int out_size, void* d_ws,
                              size_t ws_size, hipStream_t stream) {
  (void)in_sizes; (void)n_in; (void)out_size; (void)ws_size;
  // Inputs (setup_inputs order)
  const float* x      = (const float*)d_in[0];
  const float* w_in   = (const float*)d_in[1];
  const float* b_in   = (const float*)d_in[2];
  const float* w_out  = (const float*)d_in[3];
  const float* b_out  = (const float*)d_in[4];
  const float* w_glu  = (const float*)d_in[5];
  const float* w_dw   = (const float*)d_in[6];
  const float* bn_g   = (const float*)d_in[7];
  const float* bn_b   = (const float*)d_in[8];
  const float* w_pw   = (const float*)d_in[9];
  const float* w_ff1a = (const float*)d_in[10];
  const float* b_ff1a = (const float*)d_in[11];
  const float* w_ff1b = (const float*)d_in[12];
  const float* b_ff1b = (const float*)d_in[13];
  const float* w_ff2a = (const float*)d_in[14];
  const float* b_ff2a = (const float*)d_in[15];
  const float* w_ff2b = (const float*)d_in[16];
  const float* b_ff2b = (const float*)d_in[17];
  const float* ln1g = (const float*)d_in[18]; const float* ln1b = (const float*)d_in[19];
  const float* ln2g = (const float*)d_in[20]; const float* ln2b = (const float*)d_in[21];
  const float* ln3g = (const float*)d_in[22]; const float* ln3b = (const float*)d_in[23];
  const float* ln4g = (const float*)d_in[24]; const float* ln4b = (const float*)d_in[25];
  const float* ln5g = (const float*)d_in[26]; const float* ln5b = (const float*)d_in[27];

  // Workspace layout (element offsets, all 256B-aligned)
  bf16* WB = (bf16*)d_ws;
  bf16* wb_ff1a = WB;
  bf16* wb_ff1b = wb_ff1a + (size_t)CFF * CD;
  bf16* wb_ff2a = wb_ff1b + (size_t)CD * CFF;
  bf16* wb_ff2b = wb_ff2a + (size_t)CFF * CD;
  bf16* wb_in   = wb_ff2b + (size_t)CD * CFF;
  bf16* wb_out  = wb_in + (size_t)3 * CD * CD;
  bf16* wb_glu  = wb_out + (size_t)CD * CD;
  bf16* wb_pw   = wb_glu + (size_t)2 * CD * CD;
  float* Xw  = (float*)(wb_pw + (size_t)CD * CD);        // running residual [BL,D]
  float* F0  = Xw + (size_t)CBL * CD;                    // [BL,2D] scratch f32
  float* F1  = F0 + (size_t)CBL * 2 * CD;                // [BL,D] scratch f32
  bf16* Hb   = (bf16*)(F1 + (size_t)CBL * CD);           // [BL,FF] bf16 (also q/k/vT)
  bf16* Ab   = Hb + (size_t)CBL * CFF;                   // [BL,D] bf16 (LN out)
  bf16* Rb   = Ab + (size_t)CBL * CD;                    // [BL,D] bf16 (RoPE / O)
  float* stats = (float*)(Rb + (size_t)CBL * CD);        // [2*D]
  bf16* Qb = Hb;
  bf16* Kb = Hb + (size_t)CBL * CD;
  bf16* Vt = Hb + (size_t)2 * CBL * CD;
  bf16* Ob = Rb;  // RoPE buffer dead once q/k projected

  const dim3 T(256);
  auto cvt = [&](const float* s, bf16* d, int n) {
    cvt_bf16_k<<<dim3((n + 255) / 256), T, 0, stream>>>(s, d, n);
  };
  cvt(w_ff1a, wb_ff1a, CFF * CD);
  cvt(w_ff1b, wb_ff1b, CD * CFF);
  cvt(w_ff2a, wb_ff2a, CFF * CD);
  cvt(w_ff2b, wb_ff2b, CD * CFF);
  cvt(w_in, wb_in, 3 * CD * CD);
  cvt(w_out, wb_out, CD * CD);
  cvt(w_glu, wb_glu, 2 * CD * CD);
  cvt(w_pw, wb_pw, CD * CD);
  (void)hipMemcpyAsync(Xw, x, (size_t)CBL * CD * sizeof(float),
                       hipMemcpyDeviceToDevice, stream);

  const int EW = (CBL * CD) / 256;  // elementwise grid

  // ---- Stage 1: FFN1 (half-residual macaron) ----
  ln_k<0><<<CBL, T, 0, stream>>>(Xw, ln1g, ln1b, nullptr, Ab);
  gemm_bf16_k<EPI_RELU_BF16><<<dim3(CBL / 128, CFF / 64), T, 0, stream>>>(
      Ab, wb_ff1a, b_ff1a, nullptr, Hb, 1.f, CBL, CFF, CD, 0);
  gemm_bf16_k<EPI_RES><<<dim3(CBL / 128, CD / 64), T, 0, stream>>>(
      Hb, wb_ff1b, b_ff1b, Xw, nullptr, 0.5f, CBL, CD, CFF, 0);

  // ---- Stage 2: MHSA with RoPE ----
  ln_rope_k<<<CBL, T, 0, stream>>>(Xw, ln2g, ln2b, Ab, Rb);
  gemm_bf16_k<EPI_BF16><<<dim3(CBL / 128, CD / 64), T, 0, stream>>>(
      Rb, wb_in, b_in, nullptr, Qb, 1.f, CBL, CD, CD, 0);
  gemm_bf16_k<EPI_BF16><<<dim3(CBL / 128, CD / 64), T, 0, stream>>>(
      Rb, wb_in + (size_t)CD * CD, b_in + CD, nullptr, Kb, 1.f, CBL, CD, CD, 0);
  gemm_bf16_k<EPI_VT><<<dim3(CBL / 128, CD / 64), T, 0, stream>>>(
      Ab, wb_in + (size_t)2 * CD * CD, b_in + 2 * CD, nullptr, Vt, 1.f, CBL, CD, CD, CL);
  attn_k<<<dim3(CB * CH * (CL / 16)), T, 0, stream>>>(Qb, Kb, Vt, Ob);
  gemm_bf16_k<EPI_RES><<<dim3(CBL / 128, CD / 64), T, 0, stream>>>(
      Ob, wb_out, b_out, Xw, nullptr, 1.f, CBL, CD, CD, 0);

  // ---- Stage 3: convolution module ----
  ln_k<0><<<CBL, T, 0, stream>>>(Xw, ln3g, ln3b, nullptr, Ab);
  gemm_bf16_k<EPI_F32><<<dim3(CBL / 128, (2 * CD) / 64), T, 0, stream>>>(
      Ab, wb_glu, nullptr, F0, nullptr, 1.f, CBL, 2 * CD, CD, 0);
  glu_k<<<EW, T, 0, stream>>>(F0, F1);
  dwconv_k<<<EW, T, 0, stream>>>(F1, w_dw, F0);
  bnstats_k<<<CD, T, 0, stream>>>(F0, stats);
  bnsilu_k<<<EW, T, 0, stream>>>(F0, stats, bn_g, bn_b, Ab);
  gemm_bf16_k<EPI_RES><<<dim3(CBL / 128, CD / 64), T, 0, stream>>>(
      Ab, wb_pw, nullptr, Xw, nullptr, 1.f, CBL, CD, CD, 0);

  // ---- Stage 4: FFN2 + final LN ----
  ln_k<0><<<CBL, T, 0, stream>>>(Xw, ln4g, ln4b, nullptr, Ab);
  gemm_bf16_k<EPI_RELU_BF16><<<dim3(CBL / 128, CFF / 64), T, 0, stream>>>(
      Ab, wb_ff2a, b_ff2a, nullptr, Hb, 1.f, CBL, CFF, CD, 0);
  gemm_bf16_k<EPI_RES><<<dim3(CBL / 128, CD / 64), T, 0, stream>>>(
      Hb, wb_ff2b, b_ff2b, Xw, nullptr, 0.5f, CBL, CD, CFF, 0);
  ln_k<1><<<CBL, T, 0, stream>>>(Xw, ln5g, ln5b, (float*)d_out, nullptr);
}
